// UnifiedNeuroFlux_40355512713324
// MI455X (gfx1250) — matrix operand order repaired and checked by reading, not compiled, not run on hardware
//
#include <hip/hip_runtime.h>

#define Bsz 16384
#define Hd  512

typedef __attribute__((ext_vector_type(16))) __bf16 v16bf;
typedef __attribute__((ext_vector_type(8)))  float  v8f;

// ---------------------------------------------------------------- helpers ---

static __device__ __forceinline__ v8f wmma_bf16(v16bf a, v16bf b, v8f c) {
  return __builtin_amdgcn_wmma_f32_16x16x32_bf16(false, a, false, b, (short)0, c,
                                                 false, false);
}

// 16 contiguous bf16 (B fragment half-pair layout: lanes 0-15 K=0..15,
// lanes 16-31 K=16..31, 16*g already folded into the base pointer).
static __device__ __forceinline__ v16bf load16(const __bf16* p) {
  v16bf v;
#pragma unroll
  for (int e = 0; e < 16; ++e) v[e] = p[e];
  return v;
}

// A fragment (16x32 bf16). Base p = row_ptr + 16*g; element e holds
// K = kt*32 + e + 16*g  => one contiguous 16-element run.
static __device__ __forceinline__ v16bf load_a(const __bf16* p, int kt) {
  const __bf16* q = p + kt * 32;
  v16bf a;
#pragma unroll
  for (int e = 0; e < 8; ++e) a[e] = q[e];
#pragma unroll
  for (int e = 8; e < 16; ++e) a[e] = q[e];
  return a;
}

static __device__ __forceinline__ float sigmoidf_(float x) {
  return 1.0f / (1.0f + __expf(-x));
}
static __device__ __forceinline__ float softplusf_(float x) {
  return (x > 20.0f) ? x : log1pf(__expf(x));
}
static __device__ __forceinline__ float geluf_(float x) {
  float c = 0.7978845608028654f * (x + 0.044715f * x * x * x);
  return 0.5f * x * (1.0f + tanhf(c));
}

// ------------------------------------------------------------- K0: convert ---

__global__ __launch_bounds__(256) void cvt_f32_bf16(const float* __restrict__ src,
                                                    __bf16* __restrict__ dst, int n4) {
  int i = blockIdx.x * 256 + threadIdx.x;
  if (i < n4) {
    float4 v = ((const float4*)src)[i];
    dst[4 * i + 0] = (__bf16)v.x;
    dst[4 * i + 1] = (__bf16)v.y;
    dst[4 * i + 2] = (__bf16)v.z;
    dst[4 * i + 3] = (__bf16)v.w;
  }
}

// ------------------------------------------------ K1: phase-1 fused GEMMs ---
// 32 rows/block (two 16-row A sub-tiles share every B fragment), 256 threads.
// 7 GEMMs (delta,gamma,kernel,i,f,o,tanh_lin) share one A tile; nonlinear
// epilogue in registers; combined -> LDS -> LayerNorm -> ws bf16.

#define LDA (Hd + 16)

__global__ __launch_bounds__(256) void k1_phase1(
    const float* __restrict__ x, const float* __restrict__ ssm,
    const float* __restrict__ mem0, const __bf16* __restrict__ ipw,
    const float* __restrict__ ipb, const __bf16* __restrict__ kw,
    const __bf16* __restrict__ uw, const __bf16* __restrict__ mgw,
    const float* __restrict__ mscale, const float* __restrict__ lng,
    const float* __restrict__ lnb, float* __restrict__ out_ssm,
    __bf16* __restrict__ ws_comb, __bf16* __restrict__ ws_ln) {
  __shared__ __bf16 xs[32 * LDA];
  __shared__ float comb[32][Hd];
  __shared__ float redS[32][8];
  __shared__ float redQ[32][8];

  int t = threadIdx.x;
  int row0 = blockIdx.x * 32;

  // stage x tile as bf16 (coalesced float4 loads): 32 x 512
  {
    int r = t >> 3;
    int c0 = (t & 7) * 64;
    const float* p = x + (size_t)(row0 + r) * Hd + c0;
#pragma unroll
    for (int i = 0; i < 64; i += 4) {
      float4 v = *(const float4*)(p + i);
      xs[r * LDA + c0 + i + 0] = (__bf16)v.x;
      xs[r * LDA + c0 + i + 1] = (__bf16)v.y;
      xs[r * LDA + c0 + i + 2] = (__bf16)v.z;
      xs[r * LDA + c0 + i + 3] = (__bf16)v.w;
    }
  }
  __syncthreads();

  // softmax of memory_scales (M=4)
  float s0 = mscale[0], s1 = mscale[1], s2 = mscale[2], s3 = mscale[3];
  {
    float mx = fmaxf(fmaxf(s0, s1), fmaxf(s2, s3));
    s0 = __expf(s0 - mx); s1 = __expf(s1 - mx);
    s2 = __expf(s2 - mx); s3 = __expf(s3 - mx);
    float inv = 1.0f / (s0 + s1 + s2 + s3);
    s0 *= inv; s1 *= inv; s2 *= inv; s3 *= inv;
  }

  int wave = t >> 5, lane = t & 31;
  int g = lane >> 4, ml = lane & 15;
  const __bf16* pa0 = xs + (lane & 15) * LDA + 16 * g;
  const __bf16* pa1 = pa0 + 16 * LDA;

  for (int s = 0; s < 4; ++s) {
    int cb = (wave + 8 * s) * 16;
    int n = cb + ml;
    // per-lane B base pointers (column n, k contiguous, 16*g folded in)
    const __bf16* pb0 = ipw + (size_t)n * Hd + 16 * g;            // delta
    const __bf16* pb1 = ipw + (size_t)(512 + n) * Hd + 16 * g;    // gamma
    const __bf16* pb2 = kw + (size_t)n * Hd + 16 * g;             // kernel
    const __bf16* pb3 = mgw + (size_t)n * Hd + 16 * g;            // i
    const __bf16* pb4 = mgw + (size_t)(512 + n) * Hd + 16 * g;    // f
    const __bf16* pb5 = mgw + (size_t)(1024 + n) * Hd + 16 * g;   // o
    const __bf16* pb6 = uw + (size_t)n * 1536 + 16 * g;           // mem_w chunk

    v8f acc[2][7];
#pragma unroll
    for (int u = 0; u < 2; ++u)
#pragma unroll
      for (int i = 0; i < 7; ++i) acc[u][i] = (v8f){};

    for (int kt = 0; kt < 16; ++kt) {
      v16bf a0 = load_a(pa0, kt);
      v16bf a1 = load_a(pa1, kt);
      int ko = kt * 32;
      // group 1: 4 B fragments, 8 WMMAs
      {
        v16bf b0 = load16(pb0 + ko);
        v16bf b1 = load16(pb1 + ko);
        v16bf b2 = load16(pb2 + ko);
        v16bf b6 = load16(pb6 + ko);
        acc[0][0] = wmma_bf16(a0, b0, acc[0][0]);
        acc[1][0] = wmma_bf16(a1, b0, acc[1][0]);
        acc[0][1] = wmma_bf16(a0, b1, acc[0][1]);
        acc[1][1] = wmma_bf16(a1, b1, acc[1][1]);
        acc[0][2] = wmma_bf16(a0, b2, acc[0][2]);
        acc[1][2] = wmma_bf16(a1, b2, acc[1][2]);
        acc[0][6] = wmma_bf16(a0, b6, acc[0][6]);
        acc[1][6] = wmma_bf16(a1, b6, acc[1][6]);
      }
      // group 2: 3 B fragments, 6 WMMAs
      {
        v16bf b3 = load16(pb3 + ko);
        v16bf b4 = load16(pb4 + ko);
        v16bf b5 = load16(pb5 + ko);
        acc[0][3] = wmma_bf16(a0, b3, acc[0][3]);
        acc[1][3] = wmma_bf16(a1, b3, acc[1][3]);
        acc[0][4] = wmma_bf16(a0, b4, acc[0][4]);
        acc[1][4] = wmma_bf16(a1, b4, acc[1][4]);
        acc[0][5] = wmma_bf16(a0, b5, acc[0][5]);
        acc[1][5] = wmma_bf16(a1, b5, acc[1][5]);
      }
      if (kt < 14) {  // gfx1250 global_prefetch_b8 hint for k-step kt+2
        int pf = (kt + 2) * 32;
        __builtin_prefetch(pb0 + pf, 0, 3);
        __builtin_prefetch(pb1 + pf, 0, 3);
        __builtin_prefetch(pb2 + pf, 0, 3);
        __builtin_prefetch(pb3 + pf, 0, 3);
        __builtin_prefetch(pb4 + pf, 0, 3);
        __builtin_prefetch(pb5 + pf, 0, 3);
        __builtin_prefetch(pb6 + pf, 0, 3);
      }
    }

    float bi_d = ipb[n], bi_g = ipb[512 + n];
#pragma unroll
    for (int u = 0; u < 2; ++u) {
#pragma unroll
      for (int j = 0; j < 8; ++j) {
        int m = u * 16 + j + 8 * g;
        size_t b = (size_t)(row0 + m);
        size_t idx = b * Hd + n;
        float dl = softplusf_(acc[u][0][j] + bi_d);
        float gm = acc[u][1][j] + bi_g;
        float decay = __expf(-dl);
        float sn = ssm[idx] * decay + acc[u][2][j] * (1.0f - decay) * gm;
        float ig = sigmoidf_(acc[u][3][j]);
        float fg = sigmoidf_(acc[u][4][j]);
        float og = sigmoidf_(acc[u][5][j]);
        float tl = tanhf(acc[u][6][j]);
        float msum = s0 * mem0[idx] + s1 * mem0[(size_t)Bsz * Hd + idx] +
                     s2 * mem0[2ull * Bsz * Hd + idx] +
                     s3 * mem0[3ull * Bsz * Hd + idx];
        float mstate = (ig * tl + fg * msum) * og;
        out_ssm[idx] = sn;
        comb[m][n] = sn + mstate;
      }
    }
  }
  __syncthreads();

  // per-row LayerNorm over Hd, then store combined/ln as bf16 to ws
  {
    int r = t & 31, grp = t >> 5;
    int c0 = grp * 64;
    float s = 0.f, q = 0.f;
#pragma unroll
    for (int i = 0; i < 64; ++i) {
      float v = comb[r][c0 + i];
      s += v; q += v * v;
    }
    redS[r][grp] = s;
    redQ[r][grp] = q;
    __syncthreads();
    float S = 0.f, Q = 0.f;
#pragma unroll
    for (int i = 0; i < 8; ++i) { S += redS[r][i]; Q += redQ[r][i]; }
    float mean = S * (1.0f / (float)Hd);
    float var = Q * (1.0f / (float)Hd) - mean * mean;
    float rstd = rsqrtf(var + 1e-5f);
    size_t base = (size_t)(row0 + r) * Hd;
#pragma unroll
    for (int i = 0; i < 64; ++i) {
      int c = c0 + i;
      float v = comb[r][c];
      ws_comb[base + c] = (__bf16)v;
      ws_ln[base + c] = (__bf16)((v - mean) * rstd * lng[c] + lnb[c]);
    }
  }
}

// ------------------------------------------------ K2: phase-2 fused GEMMs ---
// 32 rows/block; each B fragment feeds two 16-row A sub-tiles.

__global__ __launch_bounds__(256) void k2_phase2(
    const __bf16* __restrict__ wsc, const __bf16* __restrict__ wsl,
    const __bf16* __restrict__ uw, const __bf16* __restrict__ vw1,
    const float* __restrict__ vb1, const float* __restrict__ vw2,
    const float* __restrict__ vb2, const __bf16* __restrict__ pw1,
    const float* __restrict__ pb1, const float* __restrict__ pw2,
    const float* __restrict__ pb2, float* __restrict__ out_expert,
    float* __restrict__ out_value, float* __restrict__ out_policy,
    float* __restrict__ out_ent) {
  __shared__ float rt[32][Hd];  // routing logits/T; later aliased as [vh | ph]
  __shared__ float redA[32][8];
  __shared__ float redB[32][8];
  __shared__ int ti1[32], ti2[32];
  __shared__ float tp1[32], tp2[32];

  int t = threadIdx.x, wave = t >> 5, lane = t & 31;
  int g = lane >> 4, ml = lane & 15;
  int row0 = blockIdx.x * 32;

  const __bf16* paL0 = wsl + (size_t)(row0 + (lane & 15)) * Hd + 16 * g;
  const __bf16* paL1 = paL0 + 16 * Hd;
  const __bf16* paC0 = wsc + (size_t)(row0 + (lane & 15)) * Hd + 16 * g;
  const __bf16* paC1 = paC0 + 16 * Hd;

  // --- routing GEMM: A = LN(combined), W = route chunk of unified ---
  for (int s = 0; s < 4; ++s) {
    int cb = (wave + 8 * s) * 16;
    int n = cb + ml;
    const __bf16* pb = uw + (size_t)n * 1536 + 512 + 16 * g;
    v8f a0 = {}, a1 = {};
    for (int kt = 0; kt < 16; ++kt) {
      v16bf f0 = load_a(paL0, kt);
      v16bf f1 = load_a(paL1, kt);
      v16bf b = load16(pb + kt * 32);
      a0 = wmma_bf16(f0, b, a0);
      a1 = wmma_bf16(f1, b, a1);
      if (kt < 14) __builtin_prefetch(pb + (kt + 2) * 32, 0, 3);
    }
#pragma unroll
    for (int j = 0; j < 8; ++j) {
      rt[j + 8 * g][n] = a0[j] * 10.0f;        // logits / TEMP
      rt[16 + j + 8 * g][n] = a1[j] * 10.0f;
    }
  }
  __syncthreads();

  // --- per-row softmax stats (parallel) ---
  {
    int r = t & 31, grp = t >> 5, c0 = grp * 64;
    float mx = -1e30f;
#pragma unroll
    for (int i = 0; i < 64; ++i) mx = fmaxf(mx, rt[r][c0 + i]);
    redA[r][grp] = mx;
    __syncthreads();
    float M = -1e30f;
#pragma unroll
    for (int i = 0; i < 8; ++i) M = fmaxf(M, redA[r][i]);
    float sm = 0.f;
#pragma unroll
    for (int i = 0; i < 64; ++i) sm += __expf(rt[r][c0 + i] - M);
    redB[r][grp] = sm;
  }
  __syncthreads();

  // --- top-2 scan, one lane per row ---
  if (t < 32) {
    float M = -1e30f, S = 0.f;
#pragma unroll
    for (int i = 0; i < 8; ++i) { M = fmaxf(M, redA[t][i]); S += redB[t][i]; }
    float v1 = -1e30f, v2 = -1e30f;
    int i1 = 0, i2 = 0;
    for (int c = 0; c < Hd; ++c) {
      float v = rt[t][c];
      if (v > v1) { v2 = v1; i2 = i1; v1 = v; i1 = c; }
      else if (v > v2) { v2 = v; i2 = c; }
    }
    float inv = 1.0f / S;
    float p1 = fmaxf(__expf(v1 - M) * inv, 1e-4f);
    float p2 = fmaxf(__expf(v2 - M) * inv, 1e-4f);
    float ps = p1 + p2;
    ti1[t] = i1; ti2[t] = i2;
    tp1[t] = p1 / ps; tp2[t] = p2 / ps;
  }
  __syncthreads();

  // --- expert values GEMM, gated scatter applied in epilogue ---
  for (int s = 0; s < 4; ++s) {
    int cb = (wave + 8 * s) * 16;
    int n = cb + ml;
    const __bf16* pb = uw + (size_t)n * 1536 + 1024 + 16 * g;
    v8f a0 = {}, a1 = {};
    for (int kt = 0; kt < 16; ++kt) {
      v16bf f0 = load_a(paC0, kt);
      v16bf f1 = load_a(paC1, kt);
      v16bf b = load16(pb + kt * 32);
      a0 = wmma_bf16(f0, b, a0);
      a1 = wmma_bf16(f1, b, a1);
      if (kt < 14) __builtin_prefetch(pb + (kt + 2) * 32, 0, 3);
    }
#pragma unroll
    for (int j = 0; j < 8; ++j) {
      int r = j + 8 * g;
      float d = (n == ti1[r]) ? tp1[r] : ((n == ti2[r]) ? tp2[r] : 0.0f);
      out_expert[(size_t)(row0 + r) * Hd + n] = a0[j] * d;
      int r1 = 16 + r;
      float d1 = (n == ti1[r1]) ? tp1[r1] : ((n == ti2[r1]) ? tp2[r1] : 0.0f);
      out_expert[(size_t)(row0 + r1) * Hd + n] = a1[j] * d1;
    }
  }

  // --- value / policy hidden GEMMs (256 cols each), GELU fused, into LDS ---
  for (int s = 0; s < 2; ++s) {
    int cb = (wave + 8 * s) * 16;
    int n = cb + ml;
    const __bf16* pb = vw1 + (size_t)n * Hd + 16 * g;
    v8f a0 = {}, a1 = {};
    for (int kt = 0; kt < 16; ++kt) {
      v16bf f0 = load_a(paC0, kt);
      v16bf f1 = load_a(paC1, kt);
      v16bf b = load16(pb + kt * 32);
      a0 = wmma_bf16(f0, b, a0);
      a1 = wmma_bf16(f1, b, a1);
    }
    float bb = vb1[n];
#pragma unroll
    for (int j = 0; j < 8; ++j) {
      rt[j + 8 * g][n] = geluf_(a0[j] + bb);
      rt[16 + j + 8 * g][n] = geluf_(a1[j] + bb);
    }
  }
  for (int s = 0; s < 2; ++s) {
    int cb = (wave + 8 * s) * 16;
    int n = cb + ml;
    const __bf16* pb = pw1 + (size_t)n * Hd + 16 * g;
    v8f a0 = {}, a1 = {};
    for (int kt = 0; kt < 16; ++kt) {
      v16bf f0 = load_a(paC0, kt);
      v16bf f1 = load_a(paC1, kt);
      v16bf b = load16(pb + kt * 32);
      a0 = wmma_bf16(f0, b, a0);
      a1 = wmma_bf16(f1, b, a1);
    }
    float bb = pb1[n];
#pragma unroll
    for (int j = 0; j < 8; ++j) {
      rt[j + 8 * g][256 + n] = geluf_(a0[j] + bb);
      rt[16 + j + 8 * g][256 + n] = geluf_(a1[j] + bb);
    }
  }
  __syncthreads();

  // --- tiny heads: value [B,1], policy [B,8], entropy [B] ---
  if (t < 32) {
    int r = t;
    size_t b = (size_t)(row0 + r);
    float v = vb2[0];
    for (int c = 0; c < 256; ++c) v += rt[r][c] * vw2[c];
    out_value[b] = v;
    float pl[8];
    float mx = -1e30f;
#pragma unroll
    for (int e = 0; e < 8; ++e) {
      float acc = pb2[e];
      for (int c = 0; c < 256; ++c) acc += rt[r][256 + c] * pw2[e * 256 + c];
      pl[e] = acc;
      out_policy[b * 8 + e] = acc;
      mx = fmaxf(mx, acc);
    }
    float sm = 0.f;
#pragma unroll
    for (int e = 0; e < 8; ++e) sm += __expf(pl[e] - mx);
    float lse = mx + __logf(sm);
    float ent = 0.f;
#pragma unroll
    for (int e = 0; e < 8; ++e) {
      float lp = pl[e] - lse;
      ent -= __expf(lp) * lp;
    }
    out_ent[b] = ent;
  }
}

// ----------------------------------------------------------------- launch ---

extern "C" void kernel_launch(void* const* d_in, const int* in_sizes, int n_in,
                              void* d_out, int out_size, void* d_ws, size_t ws_size,
                              hipStream_t stream) {
  (void)in_sizes; (void)n_in; (void)out_size; (void)ws_size;

  const float* x    = (const float*)d_in[0];
  const float* ssm  = (const float*)d_in[1];
  const float* mem0 = (const float*)d_in[2];
  const float* ipw  = (const float*)d_in[3];
  const float* ipb  = (const float*)d_in[4];
  const float* kw   = (const float*)d_in[5];
  const float* uw   = (const float*)d_in[6];
  const float* mgw  = (const float*)d_in[7];
  const float* msc  = (const float*)d_in[8];
  const float* lng  = (const float*)d_in[9];
  const float* lnb  = (const float*)d_in[10];
  const float* vw1  = (const float*)d_in[11];
  const float* vb1  = (const float*)d_in[12];
  const float* vw2  = (const float*)d_in[13];
  const float* vb2  = (const float*)d_in[14];
  const float* pw1  = (const float*)d_in[15];
  const float* pb1  = (const float*)d_in[16];
  const float* pw2  = (const float*)d_in[17];
  const float* pb2  = (const float*)d_in[18];

  float* out = (float*)d_out;
  float* out_expert = out;
  float* out_ssm    = out + (size_t)Bsz * Hd;
  float* out_value  = out + 2ull * Bsz * Hd;
  float* out_policy = out_value + Bsz;
  float* out_ent    = out_policy + 8ull * Bsz;

  __bf16* wsc = (__bf16*)d_ws;
  __bf16* wsl = wsc + (size_t)Bsz * Hd;
  __bf16* wp  = wsl + (size_t)Bsz * Hd;
  __bf16* b_ipw = wp; wp += 1024 * 512;
  __bf16* b_kw  = wp; wp += 512 * 512;
  __bf16* b_mgw = wp; wp += 1536 * 512;
  __bf16* b_uw  = wp; wp += 512 * 1536;
  __bf16* b_vw1 = wp; wp += 256 * 512;
  __bf16* b_pw1 = wp; wp += 256 * 512;

  auto cvt = [&](const float* s, __bf16* d, int n) {
    int n4 = n / 4;
    cvt_f32_bf16<<<(n4 + 255) / 256, 256, 0, stream>>>(s, d, n4);
  };
  cvt(ipw, b_ipw, 1024 * 512);
  cvt(kw,  b_kw,  512 * 512);
  cvt(mgw, b_mgw, 1536 * 512);
  cvt(uw,  b_uw,  512 * 1536);
  cvt(vw1, b_vw1, 256 * 512);
  cvt(pw1, b_pw1, 256 * 512);

  k1_phase1<<<Bsz / 32, 256, 0, stream>>>(x, ssm, mem0, b_ipw, ipb, b_kw, b_uw,
                                          b_mgw, msc, lng, lnb, out_ssm, wsc, wsl);
  k2_phase2<<<Bsz / 32, 256, 0, stream>>>(wsc, wsl, b_uw, b_vw1, vb1, vw2, vb2,
                                          b_pw1, pb1, pw2, pb2, out_expert,
                                          out_value, out_policy, out_ent);
}
